// EncodeProcessDecode_50044958933336
// MI455X (gfx1250) — matrix operand order repaired and checked by reading, not compiled, not run on hardware
//
#include <hip/hip_runtime.h>
#include <hip/hip_bf16.h>

// ---------------------------------------------------------------------------
// EncodeProcessDecode GNN for MI455X (gfx1250, wave32, WMMA + TDM).
//
// Precision: bf16 activations/weights, fp32 accumulation via
// v_wmma_f32_16x16x32_bf16 (16384 FLOP/inst vs 2048 for f32 WMMA).
// Activations pass ReLU+LayerNorm each layer, so bf16 storage is safe and
// halves the HBM stream (e_cat: 82 MB bf16). Node table (5 MB) and the
// segment-sum aggregate (5 MB) stay L2-resident (192 MB L2).
//
// gnn_proc_edge stages its layer-2 weight fragments (32 KB) into LDS via the
// Tensor Data Mover (tensor_load_to_lds, TENSORcnt), overlapping the DMA
// with the 24-chunk layer-1 WMMA loop and removing the redundant per-wave
// L0 streams for layer 2. This toolchain exposes the 6-arg TDM builtin.
//
// Param pointer order assumes jax pytree flattening (dicts sorted by key):
//   0 edge_attr, 1 node_attr, 2 edge_index,
//   3..8   dec_edge  (l1.W, l1.b, l2.W, l2.b, ln_b, ln_g)
//   9..14  dec_node
//   15..20 enc_edge
//   21..26 enc_node
//   27,28  out_edge (W, b)   29,30 out_node (W, b)
//   31+12i proc[i].edge (6 leaves), 37+12i proc[i].node (6 leaves)
// ---------------------------------------------------------------------------

#define GNN_N_NODES 10000
#define GNN_N_EDGES 160000
#define GNN_L 128
#define GNN_STEPS 5
#define GNN_EPS 1e-5f

typedef __bf16 bf16_t;
typedef __attribute__((ext_vector_type(16))) __bf16 v16bf;
typedef __attribute__((ext_vector_type(8))) float v8f;
typedef __attribute__((ext_vector_type(4))) unsigned int uint32x4;
typedef __attribute__((ext_vector_type(8))) int int32x8;
typedef __attribute__((ext_vector_type(4))) int int32x4;

union BF16Frag {
  v16bf v;
  uint4 q[2];
  unsigned short s[16];
};

__device__ inline unsigned short f2bf_bits(float f) {
  unsigned int u = __float_as_uint(f);
  unsigned int r = (u + 0x7fffu + ((u >> 16) & 1u)) >> 16;  // RNE
  return (unsigned short)r;
}
__device__ inline bf16_t f2bf(float f) {
  union { unsigned short s; bf16_t b; } c;
  c.s = f2bf_bits(f);
  return c.b;
}

__device__ inline v8f wmma_bf16(v16bf a, v16bf b, v8f c) {
  return __builtin_amdgcn_wmma_f32_16x16x32_bf16(
      /*neg_a=*/false, a, /*neg_b=*/false, b,
      /*c_mod=*/(short)0, c, /*reuse_a=*/false, /*reuse_b=*/false);
}

// Reduce across the 16-lane half-group that shares a C-matrix row.
__device__ inline float grp16_sum(float v) {
  v += __shfl_xor(v, 1, 16);
  v += __shfl_xor(v, 2, 16);
  v += __shfl_xor(v, 4, 16);
  v += __shfl_xor(v, 8, 16);
  return v;
}

// A fragment (16x32 bf16): lane holds row M=lane%16; lanes 0-15 take
// K=k0..k0+7 and k0+16..k0+23, lanes 16-31 take +8. koff already includes
// the 8*(lane/16) group offset. Source must be 16B aligned.
__device__ inline v16bf a_frag_from(const bf16_t* rowbase, int koff) {
  BF16Frag f;
  f.q[0] = *(const uint4*)(rowbase + koff);
  f.q[1] = *(const uint4*)(rowbase + koff + 16);
  return f.v;
}
__device__ inline v16bf a_frag_from_f32(const float* rowbase, int koff) {
  BF16Frag f;
#pragma unroll
  for (int i = 0; i < 8; ++i) f.s[i] = f2bf_bits(rowbase[koff + i]);
#pragma unroll
  for (int i = 0; i < 8; ++i) f.s[8 + i] = f2bf_bits(rowbase[koff + 16 + i]);
  return f.v;
}

// B fragment from pre-shuffled weight buffer: [kc][tile][lane][16] bf16.
// Each lane's 16 halves are contiguous (32B) -> coalesced b128 pairs.
__device__ inline v16bf b_frag(const bf16_t* wfrag, int kc, int t, int lane) {
  const bf16_t* p = wfrag + ((((kc << 3) + t) << 5) + lane) * 16;
  BF16Frag f;
  f.q[0] = *(const uint4*)p;
  f.q[1] = *(const uint4*)(p + 8);
  return f.v;
}

// ---------------------------------------------------------------------------
// Weight pre-shuffle: W[K][128] fp32 -> bf16 B-fragment order.
// idx = ((kc*8 + t)*32 + lane)*16 + h ; k = kc*32 + 8*(lane/16) + h%8 + 16*(h/8) ;
// n = t*16 + lane%16.
// ---------------------------------------------------------------------------
__global__ void gnn_prep_frag(const float* __restrict__ W,
                              bf16_t* __restrict__ out, int total) {
  for (int idx = blockIdx.x * blockDim.x + threadIdx.x; idx < total;
       idx += gridDim.x * blockDim.x) {
    int h = idx & 15;
    int lane = (idx >> 4) & 31;
    int t = (idx >> 9) & 7;
    int kc = idx >> 12;
    int k = kc * 32 + ((lane >> 4) << 3) + (h & 7) + ((h >> 3) << 4);
    int n = t * 16 + (lane & 15);
    out[idx] = f2bf(W[k * 128 + n]);
  }
}

__global__ void gnn_zero(float* __restrict__ p, int n) {
  for (int i = blockIdx.x * blockDim.x + threadIdx.x; i < n;
       i += gridDim.x * blockDim.x) p[i] = 0.0f;
}

// ---------------------------------------------------------------------------
// Encoder: x[rows][3] -> relu(x@W1+b1) -> WMMA @W2 -> relu -> LN ->
// write both halves of cat buffer ([y0 | y], identical after encoding).
// ---------------------------------------------------------------------------
__global__ __launch_bounds__(256) void gnn_encode(
    const float* __restrict__ x, int nrows,
    const float* __restrict__ W1, const float* __restrict__ b1,
    const bf16_t* __restrict__ w2frag, const float* __restrict__ b2,
    const float* __restrict__ ln_g, const float* __restrict__ ln_b,
    bf16_t* __restrict__ cat) {
  __shared__ bf16_t hsm[128][136];  // stride 136 halves -> conflict-free b128
  const int tid = threadIdx.x;
  const int rbase = blockIdx.x * 128;
  {  // layer 1: tiny K=3, scalar FMAs
    int rl = tid >> 1;
    int r = min(rbase + rl, nrows - 1);
    float x0 = x[r * 3 + 0], x1 = x[r * 3 + 1], x2 = x[r * 3 + 2];
    int c0 = (tid & 1) * 64;
#pragma unroll 8
    for (int c = c0; c < c0 + 64; ++c) {
      float v = fmaf(x0, W1[c], fmaf(x1, W1[128 + c], fmaf(x2, W1[256 + c], b1[c])));
      hsm[rl][c] = f2bf(fmaxf(v, 0.0f));
    }
  }
  __syncthreads();
  const int lane = tid & 31, w = tid >> 5;
  const int kg = lane >> 4, ln = lane & 15;
  v8f acc[8] = {};
  const bf16_t* arow = &hsm[w * 16 + ln][0];
#pragma unroll
  for (int kc = 0; kc < 4; ++kc) {
    v16bf a = a_frag_from(arow, kc * 32 + 8 * kg);
#pragma unroll
    for (int t = 0; t < 8; ++t)
      acc[t] = wmma_bf16(a, b_frag(w2frag, kc, t, lane), acc[t]);
  }
  float b2v[8], gv[8], bv[8];
#pragma unroll
  for (int t = 0; t < 8; ++t) {
    int n = t * 16 + ln;
    b2v[t] = b2[n]; gv[t] = ln_g[n]; bv[t] = ln_b[n];
  }
#pragma unroll
  for (int j = 0; j < 8; ++j) {
    float vals[8], s = 0.f, s2 = 0.f;
#pragma unroll
    for (int t = 0; t < 8; ++t) {
      float v = fmaxf(acc[t][j] + b2v[t], 0.f);
      vals[t] = v; s += v; s2 += v * v;
    }
    s = grp16_sum(s); s2 = grp16_sum(s2);
    float mu = s * (1.0f / 128.0f);
    float inv = rsqrtf(s2 * (1.0f / 128.0f) - mu * mu + GNN_EPS);
    int rstore = rbase + w * 16 + j + 8 * kg;
    if (rstore < nrows) {
      bf16_t* dst = cat + (size_t)rstore * 256;
#pragma unroll
      for (int t = 0; t < 8; ++t) {
        bf16_t y = f2bf((vals[t] - mu) * inv * gv[t] + bv[t]);
        dst[t * 16 + ln] = y;        // frozen [y0] half
        dst[128 + t * 16 + ln] = y;  // current half
      }
    }
  }
}

// ---------------------------------------------------------------------------
// Processor edge MLP, fully fused:
// X = [n_cat[recv] | n_cat[send] | e_cat[self]] (768 bf16) gathered per lane,
// 24 K-chunks of WMMA into 128-wide hidden, relu -> LDS, 4 K-chunks layer 2,
// relu + LN, write e_cat current half + atomic segment-sum into recv_agg.
//
// Dynamic LDS layout: [0, 32768)        = layer-2 weight fragments (TDM dest)
//                     [32768, 67584)    = hidden activations (128 x 136 bf16)
// The TDM load is issued by wave 0 at kernel entry and overlaps the whole
// layer-1 K loop; wave 0 drains TENSORcnt before the workgroup barrier.
// ---------------------------------------------------------------------------
#define GNN_W2LDS_BYTES 32768
#define GNN_PROC_EDGE_LDS (GNN_W2LDS_BYTES + 128 * 136 * 2)

extern __shared__ char gnn_smem[];

__global__ __launch_bounds__(256) void gnn_proc_edge(
    const bf16_t* __restrict__ ncat, bf16_t* __restrict__ ecat,
    const int* __restrict__ sendIdx, const int* __restrict__ recvIdx,
    const bf16_t* __restrict__ w1frag, const float* __restrict__ b1,
    const bf16_t* __restrict__ w2frag, const float* __restrict__ b2,
    const float* __restrict__ ln_g, const float* __restrict__ ln_b,
    float* __restrict__ recv_agg) {
  bf16_t* w2lds = (bf16_t*)gnn_smem;
  bf16_t (*hsm)[136] = (bf16_t(*)[136])(gnn_smem + GNN_W2LDS_BYTES);
  const int tid = threadIdx.x, lane = tid & 31, w = tid >> 5;
  const int kg = lane >> 4, ln = lane & 15;
  const int rbase = blockIdx.x * 128;

#if defined(__has_builtin)
#if __has_builtin(__builtin_amdgcn_tensor_load_to_lds)
  // TDM: DMA the 32 KB layer-2 weight fragment block into LDS (1-D D#:
  // count=1, type=2, data_size=4B, tensor_dim0 = tile_dim0 = 8192 dwords,
  // lds_addr = 0). One wave issues; EXEC is ignored by TENSOR ops.
  // 6-arg builtin: (g0, g1, g2, g3, g4, cpol); groups 2-4 unused (<=2D).
  if (w == 0) {
    unsigned long long ga = (unsigned long long)(uintptr_t)w2frag;
    const unsigned nd = GNN_W2LDS_BYTES / 4;  // 8192 dwords
    uint32x4 g0;
    g0[0] = 1u;                                            // count=1 (user D#)
    g0[1] = 0u;                                            // lds_addr
    g0[2] = (unsigned)(ga & 0xffffffffu);                  // global_addr lo
    g0[3] = (unsigned)((ga >> 32) & 0x01ffffffu) | (2u << 30);  // hi | type=2
    int32x8 g1;
    g1[0] = (int)(2u << 16);                     // data_size = 4 bytes
    g1[1] = (int)((nd & 0xffffu) << 16);         // tensor_dim0[15:0]
    g1[2] = (int)((nd >> 16) | (1u << 16));      // tensor_dim0[31:16] | dim1=1
    g1[3] = (int)((nd & 0xffffu) << 16);         // tile_dim0 = nd
    g1[4] = 0;                                   // tile_dim1/2 unused
    g1[5] = (int)nd;                             // tensor_dim0_stride lo
    g1[6] = 0;                                   // stride hi | dim1_stride lo
    g1[7] = 0;
    int32x4 z4 = {0, 0, 0, 0};
    int32x8 z8 = {0, 0, 0, 0, 0, 0, 0, 0};
    __builtin_amdgcn_tensor_load_to_lds(g0, g1, z4, z4, z8, 0);
  }
#else
  // Fallback: cooperative staged copy of the layer-2 weights.
  for (int i = tid; i < GNN_W2LDS_BYTES / 16; i += 256)
    ((uint4*)w2lds)[i] = ((const uint4*)w2frag)[i];
#endif
#else
  for (int i = tid; i < GNN_W2LDS_BYTES / 16; i += 256)
    ((uint4*)w2lds)[i] = ((const uint4*)w2frag)[i];
#endif

  const int r = rbase + w * 16 + ln;  // this lane's A-matrix row (edge id)
  const bf16_t* recvp = ncat + (size_t)recvIdx[r] * 256;
  const bf16_t* sendp = ncat + (size_t)sendIdx[r] * 256;
  const bf16_t* selfp = ecat + (size_t)r * 256;
  v8f acc[8] = {};
#pragma unroll
  for (int kc = 0; kc < 24; ++kc) {
    const bf16_t* src = (kc < 8) ? recvp : (kc < 16) ? sendp : selfp;
    int koff = ((kc & 7) * 32) + 8 * kg;
    __builtin_prefetch(src + koff + 64, 0, 1);  // global_prefetch_b8
    v16bf a = a_frag_from(src, koff);
#pragma unroll
    for (int t = 0; t < 8; ++t)
      acc[t] = wmma_bf16(a, b_frag(w1frag, kc, t, lane), acc[t]);
  }
  float b1v[8];
#pragma unroll
  for (int t = 0; t < 8; ++t) b1v[t] = b1[t * 16 + ln];
#pragma unroll
  for (int j = 0; j < 8; ++j) {
    int row = w * 16 + j + 8 * kg;
#pragma unroll
    for (int t = 0; t < 8; ++t)
      hsm[row][t * 16 + ln] = f2bf(fmaxf(acc[t][j] + b1v[t], 0.f));
  }
#if defined(__has_builtin)
#if __has_builtin(__builtin_amdgcn_s_wait_tensorcnt)
  if (w == 0) __builtin_amdgcn_s_wait_tensorcnt(0);  // drain TDM before barrier
#endif
#endif
  __syncthreads();
  v8f acc2[8] = {};
  const bf16_t* arow = &hsm[w * 16 + ln][0];
#pragma unroll
  for (int kc = 0; kc < 4; ++kc) {
    v16bf a = a_frag_from(arow, kc * 32 + 8 * kg);
#pragma unroll
    for (int t = 0; t < 8; ++t)
      acc2[t] = wmma_bf16(a, b_frag(w2lds, kc, t, lane), acc2[t]);
  }
  float b2v[8], gv[8], bv[8];
#pragma unroll
  for (int t = 0; t < 8; ++t) {
    int n = t * 16 + ln;
    b2v[t] = b2[n]; gv[t] = ln_g[n]; bv[t] = ln_b[n];
  }
#pragma unroll
  for (int j = 0; j < 8; ++j) {
    float vals[8], s = 0.f, s2 = 0.f;
#pragma unroll
    for (int t = 0; t < 8; ++t) {
      float v = fmaxf(acc2[t][j] + b2v[t], 0.f);
      vals[t] = v; s += v; s2 += v * v;
    }
    s = grp16_sum(s); s2 = grp16_sum(s2);
    float mu = s * (1.0f / 128.0f);
    float inv = rsqrtf(s2 * (1.0f / 128.0f) - mu * mu + GNN_EPS);
    int rstore = rbase + w * 16 + j + 8 * kg;
    bf16_t* dst = ecat + (size_t)rstore * 256 + 128;
    float* agg = recv_agg + (size_t)recvIdx[rstore] * 128;
#pragma unroll
    for (int t = 0; t < 8; ++t) {
      float y = (vals[t] - mu) * inv * gv[t] + bv[t];
      dst[t * 16 + ln] = f2bf(y);
      unsafeAtomicAdd(agg + t * 16 + ln, y);  // global_atomic_add_f32, L2-resident
    }
  }
}

// ---------------------------------------------------------------------------
// Processor node MLP: X = [n_cat[node] | recv_agg(fp32->bf16)] (384), two
// WMMA layers + LN, writes n_cat current half.
// ---------------------------------------------------------------------------
__global__ __launch_bounds__(256) void gnn_proc_node(
    bf16_t* __restrict__ ncat, const float* __restrict__ ragg,
    const bf16_t* __restrict__ w1frag, const float* __restrict__ b1,
    const bf16_t* __restrict__ w2frag, const float* __restrict__ b2,
    const float* __restrict__ ln_g, const float* __restrict__ ln_b) {
  __shared__ bf16_t hsm[128][136];
  const int tid = threadIdx.x, lane = tid & 31, w = tid >> 5;
  const int kg = lane >> 4, ln = lane & 15;
  const int rbase = blockIdx.x * 128;
  const int r = min(rbase + w * 16 + ln, GNN_N_NODES - 1);
  const bf16_t* selfp = ncat + (size_t)r * 256;
  const float* aggp = ragg + (size_t)r * 128;
  v8f acc[8] = {};
#pragma unroll
  for (int kc = 0; kc < 12; ++kc) {
    v16bf a = (kc < 8) ? a_frag_from(selfp, kc * 32 + 8 * kg)
                       : a_frag_from_f32(aggp, (kc - 8) * 32 + 8 * kg);
#pragma unroll
    for (int t = 0; t < 8; ++t)
      acc[t] = wmma_bf16(a, b_frag(w1frag, kc, t, lane), acc[t]);
  }
  float b1v[8];
#pragma unroll
  for (int t = 0; t < 8; ++t) b1v[t] = b1[t * 16 + ln];
#pragma unroll
  for (int j = 0; j < 8; ++j) {
    int row = w * 16 + j + 8 * kg;
#pragma unroll
    for (int t = 0; t < 8; ++t)
      hsm[row][t * 16 + ln] = f2bf(fmaxf(acc[t][j] + b1v[t], 0.f));
  }
  __syncthreads();
  v8f acc2[8] = {};
  const bf16_t* arow = &hsm[w * 16 + ln][0];
#pragma unroll
  for (int kc = 0; kc < 4; ++kc) {
    v16bf a = a_frag_from(arow, kc * 32 + 8 * kg);
#pragma unroll
    for (int t = 0; t < 8; ++t)
      acc2[t] = wmma_bf16(a, b_frag(w2frag, kc, t, lane), acc2[t]);
  }
  float b2v[8], gv[8], bv[8];
#pragma unroll
  for (int t = 0; t < 8; ++t) {
    int n = t * 16 + ln;
    b2v[t] = b2[n]; gv[t] = ln_g[n]; bv[t] = ln_b[n];
  }
#pragma unroll
  for (int j = 0; j < 8; ++j) {
    float vals[8], s = 0.f, s2 = 0.f;
#pragma unroll
    for (int t = 0; t < 8; ++t) {
      float v = fmaxf(acc2[t][j] + b2v[t], 0.f);
      vals[t] = v; s += v; s2 += v * v;
    }
    s = grp16_sum(s); s2 = grp16_sum(s2);
    float mu = s * (1.0f / 128.0f);
    float inv = rsqrtf(s2 * (1.0f / 128.0f) - mu * mu + GNN_EPS);
    int rstore = rbase + w * 16 + j + 8 * kg;
    if (rstore < GNN_N_NODES) {
      bf16_t* dst = ncat + (size_t)rstore * 256 + 128;
#pragma unroll
      for (int t = 0; t < 8; ++t)
        dst[t * 16 + ln] = f2bf((vals[t] - mu) * inv * gv[t] + bv[t]);
    }
  }
}

// ---------------------------------------------------------------------------
// Decoder + output head: dec MLP (128->128->128, relu, LN) then 128->1 dot.
// ---------------------------------------------------------------------------
__global__ __launch_bounds__(256) void gnn_decode(
    const bf16_t* __restrict__ cat, int nrows,
    const bf16_t* __restrict__ w1frag, const float* __restrict__ b1,
    const bf16_t* __restrict__ w2frag, const float* __restrict__ b2,
    const float* __restrict__ ln_g, const float* __restrict__ ln_b,
    const float* __restrict__ outW, const float* __restrict__ outB,
    float* __restrict__ out) {
  __shared__ bf16_t hsm[128][136];
  const int tid = threadIdx.x, lane = tid & 31, w = tid >> 5;
  const int kg = lane >> 4, ln = lane & 15;
  const int rbase = blockIdx.x * 128;
  const int r = min(rbase + w * 16 + ln, nrows - 1);
  const bf16_t* selfp = cat + (size_t)r * 256 + 128;  // current half only
  v8f acc[8] = {};
#pragma unroll
  for (int kc = 0; kc < 4; ++kc) {
    v16bf a = a_frag_from(selfp, kc * 32 + 8 * kg);
#pragma unroll
    for (int t = 0; t < 8; ++t)
      acc[t] = wmma_bf16(a, b_frag(w1frag, kc, t, lane), acc[t]);
  }
  float b1v[8];
#pragma unroll
  for (int t = 0; t < 8; ++t) b1v[t] = b1[t * 16 + ln];
#pragma unroll
  for (int j = 0; j < 8; ++j) {
    int row = w * 16 + j + 8 * kg;
#pragma unroll
    for (int t = 0; t < 8; ++t)
      hsm[row][t * 16 + ln] = f2bf(fmaxf(acc[t][j] + b1v[t], 0.f));
  }
  __syncthreads();
  v8f acc2[8] = {};
  const bf16_t* arow = &hsm[w * 16 + ln][0];
#pragma unroll
  for (int kc = 0; kc < 4; ++kc) {
    v16bf a = a_frag_from(arow, kc * 32 + 8 * kg);
#pragma unroll
    for (int t = 0; t < 8; ++t)
      acc2[t] = wmma_bf16(a, b_frag(w2frag, kc, t, lane), acc2[t]);
  }
  float b2v[8], gv[8], bv[8], wv[8];
#pragma unroll
  for (int t = 0; t < 8; ++t) {
    int n = t * 16 + ln;
    b2v[t] = b2[n]; gv[t] = ln_g[n]; bv[t] = ln_b[n]; wv[t] = outW[n];
  }
  const float ob = outB[0];
#pragma unroll
  for (int j = 0; j < 8; ++j) {
    float vals[8], s = 0.f, s2 = 0.f;
#pragma unroll
    for (int t = 0; t < 8; ++t) {
      float v = fmaxf(acc2[t][j] + b2v[t], 0.f);
      vals[t] = v; s += v; s2 += v * v;
    }
    s = grp16_sum(s); s2 = grp16_sum(s2);
    float mu = s * (1.0f / 128.0f);
    float inv = rsqrtf(s2 * (1.0f / 128.0f) - mu * mu + GNN_EPS);
    float dot = 0.f;
#pragma unroll
    for (int t = 0; t < 8; ++t)
      dot += ((vals[t] - mu) * inv * gv[t] + bv[t]) * wv[t];
    dot = grp16_sum(dot);
    int rstore = rbase + w * 16 + j + 8 * kg;
    if (ln == 0 && rstore < nrows) out[rstore] = dot + ob;
  }
}

// ---------------------------------------------------------------------------
extern "C" void kernel_launch(void* const* d_in, const int* in_sizes, int n_in,
                              void* d_out, int out_size, void* d_ws, size_t ws_size,
                              hipStream_t stream) {
  (void)in_sizes; (void)n_in; (void)out_size; (void)ws_size;
  const float* edge_attr = (const float*)d_in[0];
  const float* node_attr = (const float*)d_in[1];
  const int* eidx = (const int*)d_in[2];
  auto F = [&](int i) { return (const float*)d_in[i]; };

  // Workspace carve (~94 MB): activations + pre-shuffled bf16 weights.
  char* wp = (char*)d_ws;
  auto carve = [&](size_t bytes) {
    char* p = wp;
    wp += (bytes + 255) & ~(size_t)255;
    return p;
  };
  bf16_t* ncat = (bf16_t*)carve((size_t)GNN_N_NODES * 256 * sizeof(bf16_t));
  bf16_t* ecat = (bf16_t*)carve((size_t)GNN_N_EDGES * 256 * sizeof(bf16_t));
  float* ragg = (float*)carve((size_t)GNN_N_NODES * 128 * sizeof(float));
  auto carve_frag = [&](int K) { return (bf16_t*)carve((size_t)K * 128 * sizeof(bf16_t)); };
  bf16_t* encEw2 = carve_frag(128);
  bf16_t* encNw2 = carve_frag(128);
  bf16_t* decEw1 = carve_frag(128);
  bf16_t* decEw2 = carve_frag(128);
  bf16_t* decNw1 = carve_frag(128);
  bf16_t* decNw2 = carve_frag(128);
  bf16_t *pEw1[GNN_STEPS], *pEw2[GNN_STEPS], *pNw1[GNN_STEPS], *pNw2[GNN_STEPS];
  for (int i = 0; i < GNN_STEPS; ++i) {
    pEw1[i] = carve_frag(768);
    pEw2[i] = carve_frag(128);
    pNw1[i] = carve_frag(384);
    pNw2[i] = carve_frag(128);
  }

  auto prep = [&](const float* W, bf16_t* dst, int K) {
    int total = K * 128;
    gnn_prep_frag<<<dim3((unsigned)((total + 255) / 256)), dim3(256), 0, stream>>>(W, dst, total);
  };
  prep(F(17), encEw2, 128);
  prep(F(23), encNw2, 128);
  prep(F(3), decEw1, 128);
  prep(F(5), decEw2, 128);
  prep(F(9), decNw1, 128);
  prep(F(11), decNw2, 128);
  for (int i = 0; i < GNN_STEPS; ++i) {
    int eb = 31 + 12 * i, nb = 37 + 12 * i;
    prep(F(eb + 0), pEw1[i], 768);
    prep(F(eb + 2), pEw2[i], 128);
    prep(F(nb + 0), pNw1[i], 384);
    prep(F(nb + 2), pNw2[i], 128);
  }

  const unsigned EB = GNN_N_EDGES / 128;          // 1250
  const unsigned NB = (GNN_N_NODES + 127) / 128;  // 79
  gnn_encode<<<dim3(EB), dim3(256), 0, stream>>>(
      edge_attr, GNN_N_EDGES, F(15), F(16), encEw2, F(18), F(20), F(19), ecat);
  gnn_encode<<<dim3(NB), dim3(256), 0, stream>>>(
      node_attr, GNN_N_NODES, F(21), F(22), encNw2, F(24), F(26), F(25), ncat);

  for (int i = 0; i < GNN_STEPS; ++i) {
    int eb = 31 + 12 * i, nb = 37 + 12 * i;
    gnn_zero<<<dim3((GNN_N_NODES * 128 + 255) / 256), dim3(256), 0, stream>>>(
        ragg, GNN_N_NODES * 128);
    gnn_proc_edge<<<dim3(EB), dim3(256), GNN_PROC_EDGE_LDS, stream>>>(
        ncat, ecat, eidx, eidx + GNN_N_EDGES,
        pEw1[i], F(eb + 1), pEw2[i], F(eb + 3), F(eb + 5), F(eb + 4), ragg);
    gnn_proc_node<<<dim3(NB), dim3(256), 0, stream>>>(
        ncat, ragg, pNw1[i], F(nb + 1), pNw2[i], F(nb + 3), F(nb + 5), F(nb + 4));
  }

  float* out = (float*)d_out;
  gnn_decode<<<dim3(EB), dim3(256), 0, stream>>>(
      ecat, GNN_N_EDGES, decEw1, F(4), decEw2, F(6), F(8), F(7), F(27), F(28), out);
  gnn_decode<<<dim3(NB), dim3(256), 0, stream>>>(
      ncat, GNN_N_NODES, decNw1, F(10), decNw2, F(12), F(14), F(13), F(29), F(30),
      out + GNN_N_EDGES);
}
